// SchNet_31550829756888
// MI455X (gfx1250) — matrix operand order repaired and checked by reading, not compiled, not run on hardware
//
#include <hip/hip_runtime.h>

#define NATOMS 40000
#define NEDGES 1280000
#define NMOLS  500
#define NFEAT  128
#define NGAUSS 50

typedef __attribute__((ext_vector_type(16))) _Float16 v16h;
typedef __attribute__((ext_vector_type(8)))  float    v8f;

// shifted softplus: log(1+e^x) - log 2, numerically stable
__device__ __forceinline__ float ssp(float x) {
    float ax = fabsf(x);
    return fmaxf(x, 0.0f) + __logf(1.0f + __expf(-ax)) - 0.6931471805599453f;
}

// B fragment load: packed contiguously per (frag, lane): 16 halves = 32 bytes
__device__ __forceinline__ v16h wmma_b_frag(const _Float16* __restrict__ F, int frag, int lane) {
    return *(const v16h*)(F + ((size_t)frag * 32 + lane) * 16);
}

// Build an A fragment (16 rows x 32 K, starting at K-offset kt*32) from an f32 LDS tile.
// ISA 16-bit A layout: lanes 0-15 = rows, VGPR v<4: K=2v+s+8*group; v>=4: K=16+2(v-4)+s+8*group
__device__ __forceinline__ v16h a_frag_from_lds(const float* tile, int ldk, int kt, int lane) {
    const int group = lane >> 4;
    const int row   = lane & 15;
    v16h a;
#pragma unroll
    for (int h = 0; h < 16; ++h) {
        int v = h >> 1, s = h & 1;
        int klocal = (v < 4) ? (2 * v + s + 8 * group)
                             : (16 + 2 * (v - 4) + s + 8 * group);
        a[h] = (_Float16)tile[row * ldk + kt * 32 + klocal];
    }
    return a;
}

// Pack a KxN (row-major f32) weight into WMMA B fragments, zero-padding K up to kt*32.
// B layout (matches A): col = lane&15, K = kt*32 + 8*(lane>>4) + h for h<8, kt*32 + 16 + 8*(lane>>4) + (h-8) for h>=8
__global__ void __launch_bounds__(32)
pack_weight_kernel(const float* __restrict__ W, _Float16* __restrict__ F, int Kreal, int N) {
    const int ntiles = N >> 4;
    const int kt   = blockIdx.x / ntiles;
    const int nt   = blockIdx.x % ntiles;
    const int lane = threadIdx.x;
    const int group = lane >> 4;
    const int col   = nt * 16 + (lane & 15);
    union { _Float16 h[16]; float4 f4[2]; } u;
#pragma unroll
    for (int hh = 0; hh < 16; ++hh) {
        int K = kt * 32 + group * 8 + (hh < 8 ? hh : 16 + (hh - 8));
        u.h[hh] = (_Float16)((K < Kreal) ? W[(size_t)K * N + col] : 0.0f);
    }
    float4* dst = (float4*)(F + ((size_t)blockIdx.x * 32 + lane) * 16);
    dst[0] = u.f4[0];
    dst[1] = u.f4[1];
}

__global__ void init_kernel(const int* __restrict__ z, const float* __restrict__ embed,
                            float* __restrict__ r, float* __restrict__ agg,
                            float* __restrict__ energy) {
    size_t gid = (size_t)blockIdx.x * blockDim.x + threadIdx.x;
    if (gid < (size_t)NATOMS * NFEAT) {
        int atom = (int)(gid >> 7);
        int c    = (int)(gid & 127);
        r[gid]   = embed[(size_t)z[atom] * NFEAT + c];
        agg[gid] = 0.0f;
    }
    if (gid < NMOLS) energy[gid] = 0.0f;
}

// rf = r @ W_nf + b  (16-atom tile per wave; 4 Ktiles x 8 Ntiles WMMA)
__global__ void __launch_bounds__(32)
node_gemm_kernel(const float* __restrict__ r, const _Float16* __restrict__ Wf,
                 const float* __restrict__ bias, float* __restrict__ rf) {
    __shared__ float tile[16 * 128];
    const int lane = threadIdx.x;
    const size_t row0 = (size_t)blockIdx.x * 16;
    const float4* src = (const float4*)(r + row0 * NFEAT);
    float4* dst = (float4*)tile;
#pragma unroll
    for (int i = 0; i < 16; ++i) dst[lane + i * 32] = src[lane + i * 32];
    __syncthreads();
    v16h af[4];
#pragma unroll
    for (int kt = 0; kt < 4; ++kt) af[kt] = a_frag_from_lds(tile, 128, kt, lane);
    const int group = lane >> 4, idx = lane & 15;
#pragma unroll
    for (int nt = 0; nt < 8; ++nt) {
        v8f acc = {};
#pragma unroll
        for (int kt = 0; kt < 4; ++kt)
            acc = __builtin_amdgcn_wmma_f32_16x16x32_f16(false, af[kt], false,
                      wmma_b_frag(Wf, kt * 8 + nt, lane), (short)0, acc, false, false);
        const int col = nt * 16 + idx;
        const float bv = bias[col];
#pragma unroll
        for (int v = 0; v < 8; ++v)
            rf[(row0 + v + 8 * group) * NFEAT + col] = acc[v] + bv;
    }
}

// Fused per-16-edge tile: distances -> Gaussians -> GEMM1(ssp) -> GEMM2(ssp)
// -> gather rf -> scatter-add messages into agg (the only bulk HBM traffic).
__global__ void __launch_bounds__(32)
edge_kernel(const float* __restrict__ xyz, const int* __restrict__ nbr,
            const float* __restrict__ rf,
            const _Float16* __restrict__ Wef1, const float* __restrict__ bef1,
            const _Float16* __restrict__ Wef2, const float* __restrict__ bef2,
            float* __restrict__ agg) {
    __shared__ float sEG[16 * 64];   // Gaussians, K padded 50 -> 64
    __shared__ float sH1[16 * 128];  // hidden layer after first ssp
    __shared__ float sD[16];
    __shared__ int   sA0[16], sA1[16];
    const int lane = threadIdx.x;
    const size_t e0 = (size_t)blockIdx.x * 16;

    if (lane < 16) {
        size_t e = e0 + lane;
        int A0 = nbr[2 * e], A1 = nbr[2 * e + 1];
        sA0[lane] = A0; sA1[lane] = A1;
        float dx = xyz[3 * A0 + 0] - xyz[3 * A1 + 0];
        float dy = xyz[3 * A0 + 1] - xyz[3 * A1 + 1];
        float dz = xyz[3 * A0 + 2] - xyz[3 * A1 + 2];
        sD[lane] = sqrtf(dx * dx + dy * dy + dz * dz);
    }
    __syncthreads();

    const float GSTEP = 5.0f / 49.0f;               // CUTOFF/(N_GAUSS-1)
    const float COEFF = -0.5f / (GSTEP * GSTEP);
#pragma unroll
    for (int i = 0; i < 32; ++i) {                  // 16x64 = 1024 entries
        int j = lane + i * 32;
        int row = j >> 6, c = j & 63;
        float v = 0.0f;
        if (c < NGAUSS) { float t = sD[row] - (float)c * GSTEP; v = __expf(COEFF * t * t); }
        sEG[j] = v;
    }
    __syncthreads();

    const int group = lane >> 4, idx = lane & 15;
    v16h ag0 = a_frag_from_lds(sEG, 64, 0, lane);
    v16h ag1 = a_frag_from_lds(sEG, 64, 1, lane);
#pragma unroll
    for (int nt = 0; nt < 8; ++nt) {                // GEMM1: 16x64 @ 64x128
        v8f acc = {};
        acc = __builtin_amdgcn_wmma_f32_16x16x32_f16(false, ag0, false,
                  wmma_b_frag(Wef1, 0 * 8 + nt, lane), (short)0, acc, false, false);
        acc = __builtin_amdgcn_wmma_f32_16x16x32_f16(false, ag1, false,
                  wmma_b_frag(Wef1, 1 * 8 + nt, lane), (short)0, acc, false, false);
        const int col = nt * 16 + idx;
        const float bv = bef1[col];
#pragma unroll
        for (int v = 0; v < 8; ++v)
            sH1[(v + 8 * group) * 128 + col] = ssp(acc[v] + bv);
    }
    __syncthreads();

    v16h ah[4];
#pragma unroll
    for (int kt = 0; kt < 4; ++kt) ah[kt] = a_frag_from_lds(sH1, 128, kt, lane);
    int ea0[8], ea1[8];
#pragma unroll
    for (int v = 0; v < 8; ++v) { ea0[v] = sA0[v + 8 * group]; ea1[v] = sA1[v + 8 * group]; }

#pragma unroll
    for (int nt = 0; nt < 8; ++nt) {                // GEMM2: 16x128 @ 128x128
        v8f acc = {};
#pragma unroll
        for (int kt = 0; kt < 4; ++kt)
            acc = __builtin_amdgcn_wmma_f32_16x16x32_f16(false, ah[kt], false,
                      wmma_b_frag(Wef2, kt * 8 + nt, lane), (short)0, acc, false, false);
        const int col = nt * 16 + idx;
        const float bv = bef2[col];
#pragma unroll
        for (int v = 0; v < 8; ++v) {
            float w   = ssp(acc[v] + bv);           // edge filter value
            float rf0 = rf[(size_t)ea0[v] * NFEAT + col];
            float rf1 = rf[(size_t)ea1[v] * NFEAT + col];
            atomicAdd(&agg[(size_t)ea0[v] * NFEAT + col], rf1 * w);  // msg_ij
            atomicAdd(&agg[(size_t)ea1[v] * NFEAT + col], rf0 * w);  // msg_ji
        }
    }
}

// r += ssp(agg @ W_u1 + b) @ W_u2 + b ; also re-zeroes agg for the next layer.
__global__ void __launch_bounds__(32)
update_kernel(float* __restrict__ r, float* __restrict__ agg,
              const _Float16* __restrict__ Wu1, const float* __restrict__ bu1,
              const _Float16* __restrict__ Wu2, const float* __restrict__ bu2) {
    __shared__ float tile[16 * 128];
    const int lane = threadIdx.x;
    const size_t row0 = (size_t)blockIdx.x * 16;
    float4* asrc = (float4*)(agg + row0 * NFEAT);
    float4* dst  = (float4*)tile;
    const float4 zero4 = make_float4(0.f, 0.f, 0.f, 0.f);
#pragma unroll
    for (int i = 0; i < 16; ++i) {
        float4 v = asrc[lane + i * 32];
        dst[lane + i * 32]  = v;
        asrc[lane + i * 32] = zero4;                // reset agg for next layer/call
    }
    __syncthreads();
    v16h af[4];
#pragma unroll
    for (int kt = 0; kt < 4; ++kt) af[kt] = a_frag_from_lds(tile, 128, kt, lane);
    __syncthreads();
    const int group = lane >> 4, idx = lane & 15;
#pragma unroll
    for (int nt = 0; nt < 8; ++nt) {
        v8f acc = {};
#pragma unroll
        for (int kt = 0; kt < 4; ++kt)
            acc = __builtin_amdgcn_wmma_f32_16x16x32_f16(false, af[kt], false,
                      wmma_b_frag(Wu1, kt * 8 + nt, lane), (short)0, acc, false, false);
        const int col = nt * 16 + idx;
        const float bv = bu1[col];
#pragma unroll
        for (int v = 0; v < 8; ++v)
            tile[(v + 8 * group) * 128 + col] = ssp(acc[v] + bv);
    }
    __syncthreads();
    v16h at[4];
#pragma unroll
    for (int kt = 0; kt < 4; ++kt) at[kt] = a_frag_from_lds(tile, 128, kt, lane);
#pragma unroll
    for (int nt = 0; nt < 8; ++nt) {
        v8f acc = {};
#pragma unroll
        for (int kt = 0; kt < 4; ++kt)
            acc = __builtin_amdgcn_wmma_f32_16x16x32_f16(false, at[kt], false,
                      wmma_b_frag(Wu2, kt * 8 + nt, lane), (short)0, acc, false, false);
        const int col = nt * 16 + idx;
        const float bv = bu2[col];
#pragma unroll
        for (int v = 0; v < 8; ++v) {
            size_t o = (row0 + v + 8 * group) * NFEAT + col;
            r[o] = r[o] + acc[v] + bv;
        }
    }
}

// u = ssp(r @ W_r1 + b) @ W_r2 + b ; energy[mol] += u  (per-atom)
__global__ void __launch_bounds__(32)
readout_kernel(const float* __restrict__ r, const _Float16* __restrict__ Wr1,
               const float* __restrict__ br1, const float* __restrict__ Wr2,
               const float* __restrict__ br2, const int* __restrict__ mol,
               float* __restrict__ energy) {
    __shared__ float tile[16 * 128];
    __shared__ float uacc[16];
    const int lane = threadIdx.x;
    const size_t row0 = (size_t)blockIdx.x * 16;
    const float4* src = (const float4*)(r + row0 * NFEAT);
    float4* dst = (float4*)tile;
#pragma unroll
    for (int i = 0; i < 16; ++i) dst[lane + i * 32] = src[lane + i * 32];
    if (lane < 16) uacc[lane] = 0.0f;
    __syncthreads();
    v16h af[4];
#pragma unroll
    for (int kt = 0; kt < 4; ++kt) af[kt] = a_frag_from_lds(tile, 128, kt, lane);
    const int group = lane >> 4, idx = lane & 15;
#pragma unroll
    for (int nt = 0; nt < 4; ++nt) {                // N = 64
        v8f acc = {};
#pragma unroll
        for (int kt = 0; kt < 4; ++kt)
            acc = __builtin_amdgcn_wmma_f32_16x16x32_f16(false, af[kt], false,
                      wmma_b_frag(Wr1, kt * 4 + nt, lane), (short)0, acc, false, false);
        const int col = nt * 16 + idx;
        const float bv = br1[col];
        const float w2 = Wr2[col];
#pragma unroll
        for (int v = 0; v < 8; ++v)
            atomicAdd(&uacc[v + 8 * group], ssp(acc[v] + bv) * w2);   // ds_add_f32
    }
    __syncthreads();
    if (lane < 16) {
        size_t atom = row0 + lane;
        atomicAdd(&energy[mol[atom]], uacc[lane] + br2[0]);
    }
}

extern "C" void kernel_launch(void* const* d_in, const int* in_sizes, int n_in,
                              void* d_out, int out_size, void* d_ws, size_t ws_size,
                              hipStream_t stream) {
    (void)in_sizes; (void)n_in; (void)out_size; (void)ws_size;
    const float* xyz   = (const float*)d_in[0];
    const int*   z     = (const int*)d_in[1];
    const int*   nbr   = (const int*)d_in[2];
    const int*   mol   = (const int*)d_in[3];
    const float* embed = (const float*)d_in[4];
    const float* W_ef1 = (const float*)d_in[5];
    const float* b_ef1 = (const float*)d_in[6];
    const float* W_ef2 = (const float*)d_in[7];
    const float* b_ef2 = (const float*)d_in[8];
    const float* W_nf  = (const float*)d_in[9];
    const float* b_nf  = (const float*)d_in[10];
    const float* W_u1  = (const float*)d_in[11];
    const float* b_u1  = (const float*)d_in[12];
    const float* W_u2  = (const float*)d_in[13];
    const float* b_u2  = (const float*)d_in[14];
    const float* W_r1  = (const float*)d_in[15];
    const float* b_r1  = (const float*)d_in[16];
    const float* W_r2  = (const float*)d_in[17];
    const float* b_r2  = (const float*)d_in[18];
    float* energy = (float*)d_out;

    // workspace layout: packed f16 weight fragments, then r / rf / agg (f32, N x 128 each)
    const size_t OFF_EF1 = 0;                    // per layer 2kt*8nt*512 = 8192 halves
    const size_t OFF_EF2 = OFF_EF1 + 3 * 8192;   // per layer 4*8*512 = 16384 halves
    const size_t OFF_NF  = OFF_EF2 + 3 * 16384;
    const size_t OFF_U1  = OFF_NF  + 3 * 16384;
    const size_t OFF_U2  = OFF_U1  + 3 * 16384;
    const size_t OFF_R1  = OFF_U2  + 3 * 16384;  // 4*4*512 = 8192 halves
    const size_t HALF_BYTES = 524288;            // >= 229376 halves * 2B, aligned region

    _Float16* F   = reinterpret_cast<_Float16*>(d_ws);
    float* rbuf   = reinterpret_cast<float*>((char*)d_ws + HALF_BYTES);
    float* rfbuf  = rbuf  + (size_t)NATOMS * NFEAT;
    float* aggbuf = rfbuf + (size_t)NATOMS * NFEAT;

    for (int i = 0; i < 3; ++i) {
        pack_weight_kernel<<<16, 32, 0, stream>>>(W_ef1 + (size_t)i * NGAUSS * NFEAT, F + OFF_EF1 + i * 8192,  NGAUSS, NFEAT);
        pack_weight_kernel<<<32, 32, 0, stream>>>(W_ef2 + (size_t)i * NFEAT * NFEAT,  F + OFF_EF2 + i * 16384, NFEAT,  NFEAT);
        pack_weight_kernel<<<32, 32, 0, stream>>>(W_nf  + (size_t)i * NFEAT * NFEAT,  F + OFF_NF  + i * 16384, NFEAT,  NFEAT);
        pack_weight_kernel<<<32, 32, 0, stream>>>(W_u1  + (size_t)i * NFEAT * NFEAT,  F + OFF_U1  + i * 16384, NFEAT,  NFEAT);
        pack_weight_kernel<<<32, 32, 0, stream>>>(W_u2  + (size_t)i * NFEAT * NFEAT,  F + OFF_U2  + i * 16384, NFEAT,  NFEAT);
    }
    pack_weight_kernel<<<16, 32, 0, stream>>>(W_r1, F + OFF_R1, NFEAT, 64);

    init_kernel<<<(NATOMS * NFEAT + 255) / 256, 256, 0, stream>>>(z, embed, rbuf, aggbuf, energy);

    for (int i = 0; i < 3; ++i) {
        node_gemm_kernel<<<NATOMS / 16, 32, 0, stream>>>(rbuf, F + OFF_NF + i * 16384, b_nf + i * NFEAT, rfbuf);
        edge_kernel<<<NEDGES / 16, 32, 0, stream>>>(xyz, nbr, rfbuf,
            F + OFF_EF1 + i * 8192,  b_ef1 + i * NFEAT,
            F + OFF_EF2 + i * 16384, b_ef2 + i * NFEAT, aggbuf);
        update_kernel<<<NATOMS / 16, 32, 0, stream>>>(rbuf, aggbuf,
            F + OFF_U1 + i * 16384, b_u1 + i * NFEAT,
            F + OFF_U2 + i * 16384, b_u2 + i * NFEAT);
    }
    readout_kernel<<<NATOMS / 16, 32, 0, stream>>>(rbuf, F + OFF_R1, b_r1, W_r2, b_r2, mol, energy);
}
